// AttentionBlock_4922032521272
// MI455X (gfx1250) — compile-verified, matrix operand order checked by reading, not run on hardware
//
#include <hip/hip_runtime.h>

// ---------------------------------------------------------------------------
// AttentionBlock for MI455X (gfx1250, wave32, WMMA 16x16x32 f16 -> f32)
// B=2, C=512, H=W=64 (L=4096), 8 heads, Ch=64, GroupNorm32.
//
// Flash-attention structure (score matrix never materialized: saves ~2 GiB
// of HBM traffic). All LDS tiles are N-major/k-contiguous so every WMMA
// fragment (A and B alike) is two ds_load_b128. All global->LDS tile fills
// use the CDNA5 async path (global_load_async_to_lds_b128) with DOUBLE
// BUFFERING: tile i+1 is issued before computing tile i, and completion of
// tile i is enforced with a partial wait (s_wait_asynccnt <= in-flight of
// tile i+1), exploiting the in-order completion guarantee of ASYNCcnt.
// ---------------------------------------------------------------------------

typedef __attribute__((ext_vector_type(16))) _Float16 v16h;
typedef __attribute__((ext_vector_type(8)))  _Float16 v8h;
typedef __attribute__((ext_vector_type(8)))  float    v8f;
typedef __attribute__((ext_vector_type(4)))  float    v4f;

#define BDIM   2
#define CDIM   512
#define LDIM   4096
#define NHEAD  8
#define CH     64
#define NBH    (BDIM * NHEAD)        // 16
#define NGRP   32
#define GELEM  ((CDIM / NGRP) * LDIM)   // 65536 elems per group (contiguous)
#define ATT_SCALE 0.3535533905932738f   // 1/sqrt(sqrt(64))
#define GN_EPS 1e-5f

// ---------------------------------------------------------------------------
// CDNA5 async global->LDS copy (ASYNCcnt). LDS address = low 32 bits of the
// generic pointer (ISA 10.2: LDS aperture maps addr[31:0] to LDS_ADDR).
// ---------------------------------------------------------------------------
__device__ __forceinline__ void async_copy_b128(void* lds_dst, const void* gsrc) {
  unsigned ldsoff = (unsigned)(unsigned long long)(size_t)lds_dst;
  unsigned long long ga = (unsigned long long)(size_t)gsrc;
  asm volatile("global_load_async_to_lds_b128 %0, %1, off"
               :: "v"(ldsoff), "v"(ga) : "memory");
}
__device__ __forceinline__ void wait_async0() {
  asm volatile("s_wait_asynccnt 0x0" ::: "memory");
}
__device__ __forceinline__ void wait_async4() {   // tolerate 4 in flight
  asm volatile("s_wait_asynccnt 0x4" ::: "memory");
}
__device__ __forceinline__ void wait_async8() {   // tolerate 8 in flight
  asm volatile("s_wait_asynccnt 0x8" ::: "memory");
}

// ---------------------------------------------------------------------------
// WMMA helpers. VGPR layouts per CDNA5 ISA 7.12.2 (f16 16x16x32):
//  A: lane&15 = row M, lane>=16 selects k-half (+8);
//     elems 0..7 = K[kb..kb+7], elems 8..15 = K[kb+16..kb+23].
//  B: mirror with lane&15 = col N.
//  C/D (f32): lane&15 = col N; elem r = row (r + 8*(lane>=16)).
// With N-major (or M-major) k-contiguous tiles, both A and B fragments load
// with the same helper: two 16-byte ds_load_b128.
// ---------------------------------------------------------------------------
__device__ __forceinline__ v8f wmma16(v16h a, v16h b, v8f c) {
  return __builtin_amdgcn_wmma_f32_16x16x32_f16(false, a, false, b,
                                                (short)0, c, false, false);
}

__device__ __forceinline__ v16h frag_mk(const _Float16* base, int row0,
                                        int pitch, int k0) {
  const int lane = threadIdx.x & 31;
  const int row  = row0 + (lane & 15);
  const int kb   = k0 + ((lane & 16) ? 8 : 0);
  const _Float16* p = base + row * pitch + kb;
  v8h lo = *(const v8h*)(p);
  v8h hi = *(const v8h*)(p + 16);
  return __builtin_shufflevector(lo, hi, 0,1,2,3,4,5,6,7,
                                         8,9,10,11,12,13,14,15);
}

// ---------------------------------------------------------------------------
// GroupNorm stats: one block per (b,group); each group is a contiguous
// 65536-float slab of x.
// ---------------------------------------------------------------------------
__global__ __launch_bounds__(256) void gn_stats(const float* __restrict__ x,
                                                float* __restrict__ gstats) {
  const int bg = blockIdx.x;                       // b*32+g, 0..63
  const float* base = x + (size_t)bg * GELEM;
  float s1 = 0.f, s2 = 0.f;
  for (int i = threadIdx.x; i < GELEM / 4; i += 256) {
    v4f v = ((const v4f*)base)[i];
#pragma unroll
    for (int e = 0; e < 4; ++e) { s1 += v[e]; s2 += v[e] * v[e]; }
  }
  __shared__ float red[512];
  red[threadIdx.x] = s1;
  red[256 + threadIdx.x] = s2;
  __syncthreads();
  for (int off = 128; off > 0; off >>= 1) {
    if (threadIdx.x < off) {
      red[threadIdx.x] += red[threadIdx.x + off];
      red[256 + threadIdx.x] += red[256 + threadIdx.x + off];
    }
    __syncthreads();
  }
  if (threadIdx.x == 0) {
    float mean = red[0] / (float)GELEM;
    float var  = red[256] / (float)GELEM - mean * mean;
    gstats[bg * 2 + 0] = mean;
    gstats[bg * 2 + 1] = rsqrtf(var + GN_EPS);
  }
}

// Normalize + f16, writing the TRANSPOSE xnT[b][l][c] so the QKV GEMM's
// B-tiles are row-contiguous (k=c inner). 8 MB, lives in L2.
__global__ __launch_bounds__(256) void gn_apply(const float* __restrict__ x,
                                                const float* __restrict__ nw,
                                                const float* __restrict__ nb,
                                                const float* __restrict__ gstats,
                                                _Float16* __restrict__ xnT) {
  const size_t i4   = (size_t)blockIdx.x * 256 + threadIdx.x;  // float4 index
  const size_t base = i4 * 4;                                  // flat [b][c][l]
  const int c  = (int)((base >> 12) & (CDIM - 1));
  const int l  = (int)(base & (LDIM - 1));
  const int b  = (int)(base >> 21);
  const int bg = (int)(base >> 16);                  // b*32+g
  const float mean = gstats[bg * 2 + 0];
  const float rstd = gstats[bg * 2 + 1];
  const float w = nw[c], bb = nb[c];
  v4f v = *(const v4f*)(x + base);
#pragma unroll
  for (int e = 0; e < 4; ++e)
    xnT[((size_t)b * LDIM + l + e) * CDIM + c] =
        (_Float16)((v[e] - mean) * rstd * w + bb);
}

__global__ __launch_bounds__(256) void cvt_weights(const float* __restrict__ qkv_w,
                                                   const float* __restrict__ conv_w,
                                                   _Float16* __restrict__ wq,
                                                   _Float16* __restrict__ wp) {
  const int i = blockIdx.x * 256 + threadIdx.x;
  if (i < 3 * CDIM * CDIM) wq[i] = (_Float16)qkv_w[i];
  if (i < CDIM * CDIM)     wp[i] = (_Float16)conv_w[i];
}

// ---------------------------------------------------------------------------
// QKV GEMM: qkv[o,l] = sum_c wq[o,c] * xnT[l,c] + bias.
// 64x64 block tile, 4 waves x four 16x16 accumulators, double-buffered async
// tile pipeline. Every 64-row tile is entirely q, k or v of one head
// (192 = 3*64), so routing is uniform per block:
//   q -> qT[bh][t][c] (*scale)   k -> kT[bh][s][c] (*scale)   v -> vbuf[bh][c][s]
// ---------------------------------------------------------------------------
#define GP 40   // GEMM LDS tile pitch (32 k + pad; keeps 16B alignment)

__global__ __launch_bounds__(128) void qkv_gemm(const _Float16* __restrict__ wq,
                                                const _Float16* __restrict__ xnT,
                                                const float* __restrict__ qkv_b,
                                                _Float16* __restrict__ qT,
                                                _Float16* __restrict__ kT,
                                                _Float16* __restrict__ vbuf) {
  const int nt = blockIdx.x;          // 0..63   (l tiles)
  const int mt = blockIdx.y;          // 0..23   (o tiles)
  const int b  = blockIdx.z;          // 0..1
  __shared__ _Float16 As[2][64 * GP];  // [m=o][k=c]
  __shared__ _Float16 Bs[2][64 * GP];  // [n=l][k=c]
  const int tid  = threadIdx.x;
  const int wave = tid >> 5;
  const int lane = tid & 31;
  const int n    = lane & 15;
  const int halfsel = (lane & 16) ? 8 : 0;

  const v8f vzero = {0.f,0.f,0.f,0.f,0.f,0.f,0.f,0.f};
  v8f acc[4] = {vzero, vzero, vzero, vzero};

  const _Float16* wrow = wq  + (size_t)(mt * 64) * CDIM;
  const _Float16* xrow = xnT + ((size_t)b * LDIM + nt * 64) * CDIM;

  const int row = tid >> 1, seg = (tid & 1) * 16;   // 64 rows, 16 halfs each
  auto issue = [&](int kc, int p) {   // 4 async b128 per thread
    async_copy_b128(&As[p][row * GP + seg],     wrow + (size_t)row * CDIM + kc + seg);
    async_copy_b128(&As[p][row * GP + seg + 8], wrow + (size_t)row * CDIM + kc + seg + 8);
    async_copy_b128(&Bs[p][row * GP + seg],     xrow + (size_t)row * CDIM + kc + seg);
    async_copy_b128(&Bs[p][row * GP + seg + 8], xrow + (size_t)row * CDIM + kc + seg + 8);
  };

  issue(0, 0);
  for (int it = 0; it < CDIM / 32; ++it) {
    const int p = it & 1;
    __syncthreads();                       // readers of buf p^1 are done
    if (it + 1 < CDIM / 32) {
      issue((it + 1) * 32, p ^ 1);         // overlap next tile with compute
      wait_async4();                       // tile it complete (in-order)
    } else {
      wait_async0();
    }
    __syncthreads();                       // tile it visible to all waves
    v16h a = frag_mk(&As[p][0], wave * 16, GP, 0);
#pragma unroll
    for (int j = 0; j < 4; ++j)
      acc[j] = wmma16(a, frag_mk(&Bs[p][0], j * 16, GP, 0), acc[j]);
  }

  const int route = mt % 3;                 // 0=q 1=k 2=v
  const int h     = mt / 3;
  const int bh    = b * NHEAD + h;
  const int cbase = wave * 16 + halfsel;    // per-head channel base
  const int obase = mt * 64 + cbase;        // absolute qkv channel base
#pragma unroll
  for (int j = 0; j < 4; ++j) {
    const int l = nt * 64 + j * 16 + n;
    if (route == 0) {
      v8h o;
#pragma unroll
      for (int r = 0; r < 8; ++r)
        o[r] = (_Float16)((acc[j][r] + qkv_b[obase + r]) * ATT_SCALE);
      *(v8h*)(qT + ((size_t)bh * LDIM + l) * CH + cbase) = o;
    } else if (route == 1) {
      v8h o;
#pragma unroll
      for (int r = 0; r < 8; ++r)
        o[r] = (_Float16)((acc[j][r] + qkv_b[obase + r]) * ATT_SCALE);
      *(v8h*)(kT + ((size_t)bh * LDIM + l) * CH + cbase) = o;
    } else {
#pragma unroll
      for (int r = 0; r < 8; ++r)
        vbuf[((size_t)bh * CH + cbase + r) * LDIM + l] =
            (_Float16)(acc[j][r] + qkv_b[obase + r]);
    }
  }
}

// ---------------------------------------------------------------------------
// Flash attention. Grid (64 query tiles, 16 batch-heads), 128 threads.
// Q A-fragments live in registers for the whole pass. 64-wide K/V tiles
// stream through double-buffered LDS via the async pipeline. Online softmax
// in registers: row m of a C/D tile lives entirely in one 16-lane half, so
// shfl_xor(8,4,2,1) reduces each row. P is restaged per-wave in LDS to
// become the A-operand of O += P*V^T.
// ---------------------------------------------------------------------------
#define AT_P 72

__global__ __launch_bounds__(128) void attn_flash(const _Float16* __restrict__ qT,
                                                  const _Float16* __restrict__ kT,
                                                  const _Float16* __restrict__ vbuf,
                                                  _Float16* __restrict__ attnT) {
  const int tq = blockIdx.x;     // 0..63
  const int bh = blockIdx.y;     // 0..15
  __shared__ _Float16 Ks[2][64 * AT_P];  // [n=s][k=c]
  __shared__ _Float16 Vs[2][64 * AT_P];  // [n=c][k=s]
  __shared__ _Float16 Ps[4][16 * AT_P];  // per-wave [m=t][k=s]

  const int tid  = threadIdx.x;
  const int wave = tid >> 5;
  const int lane = tid & 31;
  const int n    = lane & 15;
  const int halfsel = (lane & 16) ? 8 : 0;

  // Q fragments: A[m=t][k=c], rows contiguous in qT.
  const _Float16* qp = qT + ((size_t)bh * LDIM + tq * 64 + wave * 16 + n) * CH;
  v16h aq0, aq1;
  {
    const int kb = halfsel;
    v8h a = *(const v8h*)(qp + kb);
    v8h bq = *(const v8h*)(qp + kb + 16);
    aq0 = __builtin_shufflevector(a, bq, 0,1,2,3,4,5,6,7,8,9,10,11,12,13,14,15);
    v8h c = *(const v8h*)(qp + 32 + kb);
    v8h d = *(const v8h*)(qp + 48 + kb);
    aq1 = __builtin_shufflevector(c, d, 0,1,2,3,4,5,6,7,8,9,10,11,12,13,14,15);
  }

  const v8f vzero = {0.f,0.f,0.f,0.f,0.f,0.f,0.f,0.f};
  v8f oacc[4] = {vzero, vzero, vzero, vzero};
  float rm[8], rl[8];
#pragma unroll
  for (int r = 0; r < 8; ++r) { rm[r] = -1e30f; rl[r] = 0.f; }

  const _Float16* kg = kT   + (size_t)bh * LDIM * CH;   // [s][c]
  const _Float16* vg = vbuf + (size_t)bh * CH * LDIM;   // [c][s]

  const int row = tid >> 1, seg = (tid & 1) * 32;       // 64 rows, 32 halfs
  auto issue = [&](int st, int p) {   // 8 async b128 per thread
    const _Float16* ksrc = kg + (size_t)(st * 64 + row) * CH + seg;
    const _Float16* vsrc = vg + (size_t)row * LDIM + st * 64 + seg;
    _Float16* kdst = &Ks[p][row * AT_P + seg];
    _Float16* vdst = &Vs[p][row * AT_P + seg];
#pragma unroll
    for (int u = 0; u < 4; ++u) {
      async_copy_b128(kdst + u * 8, ksrc + u * 8);
      async_copy_b128(vdst + u * 8, vsrc + u * 8);
    }
  };

  issue(0, 0);
  for (int st = 0; st < 64; ++st) {
    const int p = st & 1;
    __syncthreads();                       // readers of buf p^1 are done
    if (st + 1 < 64) {
      issue(st + 1, p ^ 1);                // overlap next K/V with compute
      wait_async8();                       // tile st complete (in-order)
    } else {
      wait_async0();
    }
    __syncthreads();                       // tile st visible to all waves

    // S = Q^T K  (k-dim 64 -> two wmma per 16x16 tile)
    v8f sacc[4] = {vzero, vzero, vzero, vzero};
#pragma unroll
    for (int j = 0; j < 4; ++j) {
      sacc[j] = wmma16(aq0, frag_mk(&Ks[p][0], j * 16, AT_P, 0),  sacc[j]);
      sacc[j] = wmma16(aq1, frag_mk(&Ks[p][0], j * 16, AT_P, 32), sacc[j]);
    }

    // Online softmax per query row.
    float tmax[8];
#pragma unroll
    for (int r = 0; r < 8; ++r)
      tmax[r] = fmaxf(fmaxf(sacc[0][r], sacc[1][r]),
                      fmaxf(sacc[2][r], sacc[3][r]));
#pragma unroll
    for (int off = 8; off > 0; off >>= 1)
#pragma unroll
      for (int r = 0; r < 8; ++r)
        tmax[r] = fmaxf(tmax[r], __shfl_xor(tmax[r], off, 32));

    float alpha[8];
#pragma unroll
    for (int r = 0; r < 8; ++r) {
      const float mn = fmaxf(rm[r], tmax[r]);
      alpha[r] = __expf(rm[r] - mn);
      rm[r] = mn;
      rl[r] *= alpha[r];
    }
#pragma unroll
    for (int j = 0; j < 4; ++j)
#pragma unroll
      for (int r = 0; r < 8; ++r)
        sacc[j][r] = __expf(sacc[j][r] - rm[r]);

    float tsum[8];
#pragma unroll
    for (int r = 0; r < 8; ++r)
      tsum[r] = sacc[0][r] + sacc[1][r] + sacc[2][r] + sacc[3][r];
#pragma unroll
    for (int off = 8; off > 0; off >>= 1)
#pragma unroll
      for (int r = 0; r < 8; ++r)
        tsum[r] += __shfl_xor(tsum[r], off, 32);
#pragma unroll
    for (int r = 0; r < 8; ++r) rl[r] += tsum[r];

#pragma unroll
    for (int j = 0; j < 4; ++j)
#pragma unroll
      for (int r = 0; r < 8; ++r) oacc[j][r] *= alpha[r];

    // Restage P (per-wave LDS region; same-wave DS ops are ordered).
    _Float16* pw = &Ps[wave][0];
#pragma unroll
    for (int j = 0; j < 4; ++j)
#pragma unroll
      for (int r = 0; r < 8; ++r)
        pw[(r + halfsel) * AT_P + j * 16 + n] = (_Float16)sacc[j][r];

    v16h pa0 = frag_mk(pw, 0, AT_P, 0);
    v16h pa1 = frag_mk(pw, 0, AT_P, 32);
#pragma unroll
    for (int j = 0; j < 4; ++j) {
      oacc[j] = wmma16(pa0, frag_mk(&Vs[p][0], j * 16, AT_P, 0),  oacc[j]);
      oacc[j] = wmma16(pa1, frag_mk(&Vs[p][0], j * 16, AT_P, 32), oacc[j]);
    }
  }

  // attnT[b][t][h*64 + c] so the projection GEMM B-tiles are row-contiguous.
  const int b   = bh >> 3;
  const int cg0 = (bh & 7) * CH;
  const int t0  = tq * 64 + wave * 16 + halfsel;
#pragma unroll
  for (int j = 0; j < 4; ++j) {
    const int cg = cg0 + j * 16 + n;
#pragma unroll
    for (int r = 0; r < 8; ++r)
      attnT[((size_t)b * LDIM + t0 + r) * CDIM + cg] =
          (_Float16)(oacc[j][r] / rl[r]);
  }
}

// ---------------------------------------------------------------------------
// Projection GEMM + bias + residual: out = x + conv_w @ attn + conv_b.
// Same double-buffered tiling as qkv_gemm; B comes from attnT (N-major).
// ---------------------------------------------------------------------------
__global__ __launch_bounds__(128) void proj_gemm(const _Float16* __restrict__ wp,
                                                 const _Float16* __restrict__ attnT,
                                                 const float* __restrict__ conv_b,
                                                 const float* __restrict__ x,
                                                 float* __restrict__ out) {
  const int nt = blockIdx.x;          // 0..63
  const int mt = blockIdx.y;          // 0..7
  const int b  = blockIdx.z;          // 0..1
  __shared__ _Float16 As[2][64 * GP];
  __shared__ _Float16 Bs[2][64 * GP];
  const int tid  = threadIdx.x;
  const int wave = tid >> 5;
  const int lane = tid & 31;
  const int n    = lane & 15;
  const int halfsel = (lane & 16) ? 8 : 0;

  const v8f vzero = {0.f,0.f,0.f,0.f,0.f,0.f,0.f,0.f};
  v8f acc[4] = {vzero, vzero, vzero, vzero};

  const _Float16* wrow = wp + (size_t)(mt * 64) * CDIM;
  const _Float16* arow = attnT + ((size_t)b * LDIM + nt * 64) * CDIM;

  const int row = tid >> 1, seg = (tid & 1) * 16;
  auto issue = [&](int kc, int p) {
    async_copy_b128(&As[p][row * GP + seg],     wrow + (size_t)row * CDIM + kc + seg);
    async_copy_b128(&As[p][row * GP + seg + 8], wrow + (size_t)row * CDIM + kc + seg + 8);
    async_copy_b128(&Bs[p][row * GP + seg],     arow + (size_t)row * CDIM + kc + seg);
    async_copy_b128(&Bs[p][row * GP + seg + 8], arow + (size_t)row * CDIM + kc + seg + 8);
  };

  issue(0, 0);
  for (int it = 0; it < CDIM / 32; ++it) {
    const int p = it & 1;
    __syncthreads();
    if (it + 1 < CDIM / 32) {
      issue((it + 1) * 32, p ^ 1);
      wait_async4();
    } else {
      wait_async0();
    }
    __syncthreads();
    v16h a = frag_mk(&As[p][0], wave * 16, GP, 0);
#pragma unroll
    for (int j = 0; j < 4; ++j)
      acc[j] = wmma16(a, frag_mk(&Bs[p][0], j * 16, GP, 0), acc[j]);
  }

#pragma unroll
  for (int j = 0; j < 4; ++j) {
    const int l = nt * 64 + j * 16 + n;
#pragma unroll
    for (int r = 0; r < 8; ++r) {
      const int o = mt * 64 + wave * 16 + halfsel + r;
      const size_t idx = ((size_t)b * CDIM + o) * LDIM + l;
      out[idx] = acc[j][r] + conv_b[o] + x[idx];
    }
  }
}

// ---------------------------------------------------------------------------
// Workspace layout (bytes):
//   xnT   : 0       8 MB  f16 [b][l][c]
//   qT    : 8 MB    8 MB  f16 [bh][t][c] (pre-scaled)
//   kT    : 16 MB   8 MB  f16 [bh][s][c] (pre-scaled)
//   vbuf  : 24 MB   8 MB  f16 [bh][c][s]
//   attnT : 32 MB   8 MB  f16 [b][t][c]
//   wq_h  : 40 MB   1.5 MB
//   wp_h  : 41.5 MB 0.5 MB
//   gstats: 42 MB   512 B
// ---------------------------------------------------------------------------
extern "C" void kernel_launch(void* const* d_in, const int* in_sizes, int n_in,
                              void* d_out, int out_size, void* d_ws, size_t ws_size,
                              hipStream_t stream) {
  const float* x      = (const float*)d_in[0];
  const float* norm_w = (const float*)d_in[1];
  const float* norm_b = (const float*)d_in[2];
  const float* qkv_w  = (const float*)d_in[3];
  const float* qkv_b  = (const float*)d_in[4];
  const float* conv_w = (const float*)d_in[5];
  const float* conv_b = (const float*)d_in[6];
  float* out = (float*)d_out;

  char* ws = (char*)d_ws;
  const size_t SZ8 = (size_t)BDIM * CDIM * LDIM * sizeof(_Float16);  // 8 MB
  _Float16* xnT  = (_Float16*)(ws);
  _Float16* qT   = (_Float16*)(ws + SZ8);
  _Float16* kT   = (_Float16*)(ws + 2 * SZ8);
  _Float16* vbuf = (_Float16*)(ws + 3 * SZ8);
  _Float16* attn = (_Float16*)(ws + 4 * SZ8);
  _Float16* wq_h = (_Float16*)(ws + 5 * SZ8);
  _Float16* wp_h = (_Float16*)(ws + 5 * SZ8 + (size_t)3 * CDIM * CDIM * 2);
  float*    gst  = (float*)   (ws + 5 * SZ8 + (size_t)4 * CDIM * CDIM * 2);

  gn_stats<<<dim3(BDIM * NGRP), dim3(256), 0, stream>>>(x, gst);
  gn_apply<<<dim3((BDIM * CDIM * LDIM) / (4 * 256)), dim3(256), 0, stream>>>(
      x, norm_w, norm_b, gst, xnT);
  cvt_weights<<<dim3((3 * CDIM * CDIM) / 256), dim3(256), 0, stream>>>(
      qkv_w, conv_w, wq_h, wp_h);
  qkv_gemm<<<dim3(LDIM / 64, (3 * CDIM) / 64, BDIM), dim3(128), 0, stream>>>(
      wq_h, xnT, qkv_b, qT, kT, vbuf);
  attn_flash<<<dim3(LDIM / 64, NBH), dim3(128), 0, stream>>>(qT, kT, vbuf, attn);
  proj_gemm<<<dim3(LDIM / 64, CDIM / 64, BDIM), dim3(128), 0, stream>>>(
      wp_h, attn, conv_b, x, out);
}
